// QuantizedLinear_50079318671750
// MI455X (gfx1250) — compile-verified, matrix operand order checked by reading, not run on hardware
//
#include <hip/hip_runtime.h>

#define IN_F   4096
#define OUT_F  4096
#define GROUPS 64      // quant groups per out-feature
#define PK_ROW 32      // int32 elements per packed group row
#define BM 128
#define BN 128
#define BK 32
#define LDS_STRIDE 40  // bf16 elems per LDS row: 32 data + 8 pad (80B, 16B aligned)
#define NKC (IN_F / BK)

typedef __attribute__((ext_vector_type(16))) __bf16 v16bf;
typedef __attribute__((ext_vector_type(8)))  float  v8f;
typedef __attribute__((ext_vector_type(4)))  int    v4i_t;
typedef unsigned short ushort_t;

#if __has_builtin(__builtin_amdgcn_global_load_async_to_lds_b128) && \
    __has_builtin(__builtin_amdgcn_s_wait_asynccnt)
#define HAVE_ASYNC 1
#else
#define HAVE_ASYNC 0
#endif

#define AS1 __attribute__((address_space(1)))
#define AS3 __attribute__((address_space(3)))

union FragBF { v16bf v; uint4 q[2]; };

__device__ __forceinline__ ushort_t f2bf(float f) {
  unsigned u = __float_as_uint(f);
  u += 0x7FFFu + ((u >> 16) & 1u);     // round-to-nearest-even
  return (ushort_t)(u >> 16);
}

// ---------- pass 1a: x (f32) -> bf16, done once ----------
__global__ __launch_bounds__(256)
void convert_x_bf16(const float* __restrict__ x, ushort_t* __restrict__ xb) {
  const size_t i = ((size_t)blockIdx.x * 256 + threadIdx.x) * 8;
  const float4* p = reinterpret_cast<const float4*>(x + i);
  float4 a = p[0], b = p[1];
  ushort_t tmp[8] = {f2bf(a.x), f2bf(a.y), f2bf(a.z), f2bf(a.w),
                     f2bf(b.x), f2bf(b.y), f2bf(b.z), f2bf(b.w)};
  *reinterpret_cast<uint4*>(xb + i) = *reinterpret_cast<uint4*>(tmp);
}

// ---------- pass 1b: 4-bit grouped weights -> bf16 [O][K], done once ----------
__global__ __launch_bounds__(256)
void dequant_w_bf16(const int* __restrict__ wpk, const float* __restrict__ scales,
                    const float* __restrict__ zeros, ushort_t* __restrict__ wb) {
  const int u    = blockIdx.x * 256 + threadIdx.x;   // 0 .. O*G*2-1
  const int pair = u >> 1;                           // o*64 + g
  const int half = u & 1;                            // which 16 ints of the group row
  const float s = scales[pair], z = zeros[pair];
  const int4* p = reinterpret_cast<const int4*>(wpk + pair * PK_ROW + half * 16);
  ushort_t tmp[32];
#pragma unroll
  for (int q = 0; q < 4; ++q) {
    const int4 v = p[q];
    const int vals[4] = {v.x, v.y, v.z, v.w};
#pragma unroll
    for (int j = 0; j < 4; ++j) {
      const int b = vals[j];
      const int idx = q * 8 + j * 2;
      tmp[idx]     = f2bf((float)(b & 15)        * s + z);  // low nibble -> even slot
      tmp[idx + 1] = f2bf((float)((b >> 4) & 15) * s + z);  // high nibble -> odd slot
    }
  }
  const int o = pair >> 6, g = pair & 63;
  uint4* d4 = reinterpret_cast<uint4*>(wb + (size_t)o * IN_F + g * 64 + half * 32);
  const uint4* s4 = reinterpret_cast<const uint4*>(tmp);
  d4[0] = s4[0]; d4[1] = s4[1]; d4[2] = s4[2]; d4[3] = s4[3];
}

// ---------- pass 2: pure bf16 GEMM + bias ----------
__global__ __launch_bounds__(256)
void gemm_bf16(const ushort_t* __restrict__ xa, const ushort_t* __restrict__ wb,
               const float* __restrict__ bias, float* __restrict__ out)
{
  const int t    = threadIdx.x;
  const int lane = t & 31;
  const int wave = t >> 5;
  const int l15  = lane & 15;
  const int hi   = lane >> 4;
  const int wm   = wave & 3;
  const int wn   = wave >> 2;
  const int bm0  = blockIdx.y * BM;
  const int bn0  = blockIdx.x * BN;

  const int arow  = t >> 1;          // 0..127
  const int akseg = (t & 1) * 16;    // k-local 0 or 16 (32B per thread)
  const int bcol  = t >> 1;
  const int bkseg = (t & 1) * 16;

  const ushort_t* agp = xa + (size_t)(bm0 + arow) * IN_F + akseg;
  const ushort_t* bgp = wb + (size_t)(bn0 + bcol) * IN_F + bkseg;

  v8f acc[2][4] = {};

#if HAVE_ASYNC
  // double-buffered LDS, copies tracked by ASYNCcnt
  __shared__ __align__(16) ushort_t As[2][BM * LDS_STRIDE];
  __shared__ __align__(16) ushort_t Bs[2][BN * LDS_STRIDE];

  auto issue = [&](int buf, int k0) {
    AS1 v4i_t* ga = (AS1 v4i_t*)const_cast<ushort_t*>(agp + k0);
    AS1 v4i_t* gb = (AS1 v4i_t*)const_cast<ushort_t*>(bgp + k0);
    AS3 v4i_t* la = (AS3 v4i_t*)&As[buf][arow * LDS_STRIDE + akseg];
    AS3 v4i_t* lb = (AS3 v4i_t*)&Bs[buf][bcol * LDS_STRIDE + bkseg];
    // IOFFSET is added to both global and LDS address (ISA §10.2 async pseudocode)
    __builtin_amdgcn_global_load_async_to_lds_b128(ga, la, 0,  0);
    __builtin_amdgcn_global_load_async_to_lds_b128(ga, la, 16, 0);
    __builtin_amdgcn_global_load_async_to_lds_b128(gb, lb, 0,  0);
    __builtin_amdgcn_global_load_async_to_lds_b128(gb, lb, 16, 0);
  };

  issue(0, 0);
  int pbuf = 0;
  for (int kc = 0; kc < NKC; ++kc) {
    __syncthreads();                       // other buffer no longer being read
    const int knext = (kc + 1 < NKC) ? (kc + 1) * BK : 0;   // last iter: dummy refill
    issue(pbuf ^ 1, knext);
    __builtin_amdgcn_s_wait_asynccnt(4);   // current buffer's 4 copies done (in-order)
    __syncthreads();                       // publish to all waves

    FragBF a[2], b[4];
#pragma unroll
    for (int mt = 0; mt < 2; ++mt) {
      const ushort_t* src = &As[pbuf][(wm * 32 + mt * 16 + l15) * LDS_STRIDE + hi * 8];
      a[mt].q[0] = *reinterpret_cast<const uint4*>(src);
      a[mt].q[1] = *reinterpret_cast<const uint4*>(src + 16);
    }
#pragma unroll
    for (int nt = 0; nt < 4; ++nt) {
      const ushort_t* src = &Bs[pbuf][(wn * 64 + nt * 16 + l15) * LDS_STRIDE + hi * 8];
      b[nt].q[0] = *reinterpret_cast<const uint4*>(src);
      b[nt].q[1] = *reinterpret_cast<const uint4*>(src + 16);
    }
#pragma unroll
    for (int mt = 0; mt < 2; ++mt)
#pragma unroll
      for (int nt = 0; nt < 4; ++nt)
        acc[mt][nt] = __builtin_amdgcn_wmma_f32_16x16x32_bf16(
            false, a[mt].v, false, b[nt].v, (short)0, acc[mt][nt], false, false);
    pbuf ^= 1;
  }
#else
  // sync fallback: register prefetch pipeline, single LDS buffer
  __shared__ __align__(16) ushort_t As[BM * LDS_STRIDE];
  __shared__ __align__(16) ushort_t Bs[BN * LDS_STRIDE];

  uint4 aq[2], bq[2];
  auto loadRegs = [&](int k0) {
    const uint4* pa = reinterpret_cast<const uint4*>(agp + k0);
    const uint4* pb = reinterpret_cast<const uint4*>(bgp + k0);
    aq[0] = pa[0]; aq[1] = pa[1];
    bq[0] = pb[0]; bq[1] = pb[1];
  };
  loadRegs(0);
  for (int kc = 0; kc < NKC; ++kc) {
    __syncthreads();
    {
      uint4* da = reinterpret_cast<uint4*>(&As[arow * LDS_STRIDE + akseg]);
      uint4* db = reinterpret_cast<uint4*>(&Bs[bcol * LDS_STRIDE + bkseg]);
      da[0] = aq[0]; da[1] = aq[1];
      db[0] = bq[0]; db[1] = bq[1];
    }
    __syncthreads();
    if (kc + 1 < NKC) loadRegs((kc + 1) * BK);

    FragBF a[2], b[4];
#pragma unroll
    for (int mt = 0; mt < 2; ++mt) {
      const ushort_t* src = &As[(wm * 32 + mt * 16 + l15) * LDS_STRIDE + hi * 8];
      a[mt].q[0] = *reinterpret_cast<const uint4*>(src);
      a[mt].q[1] = *reinterpret_cast<const uint4*>(src + 16);
    }
#pragma unroll
    for (int nt = 0; nt < 4; ++nt) {
      const ushort_t* src = &Bs[(wn * 64 + nt * 16 + l15) * LDS_STRIDE + hi * 8];
      b[nt].q[0] = *reinterpret_cast<const uint4*>(src);
      b[nt].q[1] = *reinterpret_cast<const uint4*>(src + 16);
    }
#pragma unroll
    for (int mt = 0; mt < 2; ++mt)
#pragma unroll
      for (int nt = 0; nt < 4; ++nt)
        acc[mt][nt] = __builtin_amdgcn_wmma_f32_16x16x32_bf16(
            false, a[mt].v, false, b[nt].v, (short)0, acc[mt][nt], false, false);
  }
#endif

  // epilogue: bias + coalesced stores
#pragma unroll
  for (int nt = 0; nt < 4; ++nt) {
    const int col = bn0 + wn * 64 + nt * 16 + l15;
    const float bv = bias[col];
#pragma unroll
    for (int mt = 0; mt < 2; ++mt) {
      const int rbase = bm0 + wm * 32 + mt * 16 + hi * 8;
#pragma unroll
      for (int r = 0; r < 8; ++r)
        out[(size_t)(rbase + r) * OUT_F + col] = acc[mt][nt][r] + bv;
    }
  }
}

// ---------- fallback: fused single-pass kernel (if ws too small) ----------
__global__ __launch_bounds__(256)
void qlinear_fused(const float* __restrict__ x, const int* __restrict__ wpk,
                   const float* __restrict__ scales, const float* __restrict__ zeros,
                   const float* __restrict__ bias, float* __restrict__ out)
{
  __shared__ __align__(16) ushort_t As[BM * LDS_STRIDE];
  __shared__ __align__(16) ushort_t Bs[BN * LDS_STRIDE];

  const int t = threadIdx.x;
  const int lane = t & 31, wave = t >> 5;
  const int l15 = lane & 15, hi = lane >> 4;
  const int wm = wave & 3, wn = wave >> 2;
  const int bm0 = blockIdx.y * BM, bn0 = blockIdx.x * BN;
  const int arow = t >> 1, akseg = (t & 1) * 16;
  const int bcol = t >> 1, bhalf = t & 1;
  const float* xrow = x + (size_t)(bm0 + arow) * IN_F + akseg;

  float4 af[4]; int4 pw[2]; float ws, wz;
  auto loadA = [&](int k0) {
    const float4* p = reinterpret_cast<const float4*>(xrow + k0);
    af[0] = p[0]; af[1] = p[1]; af[2] = p[2]; af[3] = p[3];
  };
  auto loadB = [&](int k0) {
    const int g = k0 >> 6;
    const int row = (bn0 + bcol) * GROUPS + g;
    const int base = row * PK_ROW + ((k0 & 63) >> 1) + bhalf * 8;
    const int4* p = reinterpret_cast<const int4*>(wpk + base);
    pw[0] = p[0]; pw[1] = p[1];
    ws = scales[row]; wz = zeros[row];
  };

  v8f acc[2][4] = {};
  loadA(0); loadB(0);
  for (int kc = 0; kc < NKC; ++kc) {
    __syncthreads();
    {
      ushort_t tmp[16];
      const float* f = reinterpret_cast<const float*>(af);
#pragma unroll
      for (int i = 0; i < 16; ++i) tmp[i] = f2bf(f[i]);
      uint4* dst = reinterpret_cast<uint4*>(&As[arow * LDS_STRIDE + akseg]);
      dst[0] = *reinterpret_cast<uint4*>(&tmp[0]);
      dst[1] = *reinterpret_cast<uint4*>(&tmp[8]);
    }
    {
      ushort_t tmp[16];
      const int* p = reinterpret_cast<const int*>(pw);
#pragma unroll
      for (int j = 0; j < 8; ++j) {
        const int b = p[j];
        tmp[2 * j]     = f2bf((float)(b & 15)        * ws + wz);
        tmp[2 * j + 1] = f2bf((float)((b >> 4) & 15) * ws + wz);
      }
      uint4* dst = reinterpret_cast<uint4*>(&Bs[bcol * LDS_STRIDE + bhalf * 16]);
      dst[0] = *reinterpret_cast<uint4*>(&tmp[0]);
      dst[1] = *reinterpret_cast<uint4*>(&tmp[8]);
    }
    __syncthreads();
    if (kc + 1 < NKC) { loadA((kc + 1) * BK); loadB((kc + 1) * BK); }

    FragBF a[2], b[4];
#pragma unroll
    for (int mt = 0; mt < 2; ++mt) {
      const ushort_t* src = &As[(wm * 32 + mt * 16 + l15) * LDS_STRIDE + hi * 8];
      a[mt].q[0] = *reinterpret_cast<const uint4*>(src);
      a[mt].q[1] = *reinterpret_cast<const uint4*>(src + 16);
    }
#pragma unroll
    for (int nt = 0; nt < 4; ++nt) {
      const ushort_t* src = &Bs[(wn * 64 + nt * 16 + l15) * LDS_STRIDE + hi * 8];
      b[nt].q[0] = *reinterpret_cast<const uint4*>(src);
      b[nt].q[1] = *reinterpret_cast<const uint4*>(src + 16);
    }
#pragma unroll
    for (int mt = 0; mt < 2; ++mt)
#pragma unroll
      for (int nt = 0; nt < 4; ++nt)
        acc[mt][nt] = __builtin_amdgcn_wmma_f32_16x16x32_bf16(
            false, a[mt].v, false, b[nt].v, (short)0, acc[mt][nt], false, false);
  }

#pragma unroll
  for (int nt = 0; nt < 4; ++nt) {
    const int col = bn0 + wn * 64 + nt * 16 + l15;
    const float bv = bias[col];
#pragma unroll
    for (int mt = 0; mt < 2; ++mt) {
      const int rbase = bm0 + wm * 32 + mt * 16 + hi * 8;
#pragma unroll
      for (int r = 0; r < 8; ++r)
        out[(size_t)(rbase + r) * OUT_F + col] = acc[mt][nt][r] + bv;
    }
  }
}

extern "C" void kernel_launch(void* const* d_in, const int* in_sizes, int n_in,
                              void* d_out, int out_size, void* d_ws, size_t ws_size,
                              hipStream_t stream) {
  (void)n_in; (void)out_size;
  const float* x   = (const float*)d_in[0];
  const int*   wpk = (const int*)d_in[1];
  const float* sc  = (const float*)d_in[2];
  const float* zr  = (const float*)d_in[3];
  const float* bs  = (const float*)d_in[4];
  float* out = (float*)d_out;

  const int M = in_sizes[0] / IN_F;                       // 8192
  const size_t needA = (size_t)M * IN_F * sizeof(ushort_t);      // 67 MB
  const size_t needW = (size_t)OUT_F * IN_F * sizeof(ushort_t);  // 33.5 MB

  if (d_ws && ws_size >= needA + needW) {
    ushort_t* xb = (ushort_t*)d_ws;
    ushort_t* wb = (ushort_t*)((char*)d_ws + needA);
    const long long nx = (long long)M * IN_F;
    convert_x_bf16<<<(unsigned)(nx / (256 * 8)), 256, 0, stream>>>(x, xb);
    dequant_w_bf16<<<(OUT_F * GROUPS * 2) / 256, 256, 0, stream>>>(wpk, sc, zr, wb);
    gemm_bf16<<<dim3(OUT_F / BN, M / BM), 256, 0, stream>>>(xb, wb, bs, out);
  } else {
    qlinear_fused<<<dim3(OUT_F / BN, M / BM), 256, 0, stream>>>(x, wpk, sc, zr, bs, out);
  }
}
